// EfficientMixedAttention_86414741996167
// MI455X (gfx1250) — matrix operand order repaired and checked by reading, not compiled, not run on hardware
//
#include <hip/hip_runtime.h>

typedef __attribute__((ext_vector_type(16))) __bf16 v16bf;
typedef __attribute__((ext_vector_type(8)))  float  v8f;

union FragU { v16bf v; unsigned int u[8]; };

__device__ __forceinline__ unsigned short f2bf(float f) {
  unsigned int u = __float_as_uint(f);
  u += 0x7FFFu + ((u >> 16) & 1u);          // round-to-nearest-even
  return (unsigned short)(u >> 16);
}

// A fragment: 16x32 bf16, lane L: m = base + L%16, c = L/16
//   element i<8 : K = 8c + i ; element i>=8 : K = 16 + 8c + (i-8)
__device__ __forceinline__ v16bf load_a_frag(const unsigned short* base, long ld,
                                             int m, int k0, int lane) {
  int mm = m + (lane & 15);
  int c  = lane >> 4;
  const unsigned int* row = (const unsigned int*)(base + (long)mm * ld + k0);
  FragU f;
#pragma unroll
  for (int j = 0; j < 4; ++j) f.u[j]     = row[4 * c + j];      // K = 8c+2j
#pragma unroll
  for (int j = 0; j < 4; ++j) f.u[4 + j] = row[8 + 4 * c + j];  // K = 16+8c+2j
  return f.v;
}

// B fragment from transposed operand Bt[n][k]: lane L: n = base + L%16, c=L/16
//   element i<8 : K = 8c + i ; element i>=8 : K = 16 + 8c + (i-8)
__device__ __forceinline__ v16bf load_b_frag(const unsigned short* base, long ld,
                                             int n, int k0, int lane) {
  int nn = n + (lane & 15);
  int c  = lane >> 4;
  const unsigned int* row = (const unsigned int*)(base + (long)nn * ld + k0);
  FragU f;
#pragma unroll
  for (int j = 0; j < 8; ++j) f.u[j] = row[(j < 4 ? 0 : 4) + 4 * c + j];  // K = 8c+2j | 16+8c+2(j-4)
  return f.v;
}

__device__ __forceinline__ v8f wmma_bf16(v16bf a, v16bf b, v8f c) {
  return __builtin_amdgcn_wmma_f32_16x16x32_bf16(false, a, false, b, (short)0, c,
                                                 false, false);
}

// CDNA5 async global->LDS copy (16B per lane), tracked by ASYNCcnt.
// lds addr VGPR = low 32 bits of flat shared pointer (LDS aperture offset).
__device__ __forceinline__ void async_b128(void* lds, const void* gaddr) {
  unsigned ldsa = (unsigned)(size_t)lds;
  unsigned long long ga = (unsigned long long)(size_t)gaddr;
  asm volatile("global_load_async_to_lds_b128 %0, %1, off"
               :: "v"(ldsa), "v"(ga) : "memory");
}
__device__ __forceinline__ void wait_async0() {
  asm volatile("s_wait_asynccnt 0x0" ::: "memory");
}

// ---------------------------------------------------------------------------
// Weight transpose + f32->bf16:  Wt[n*K + k] = bf16(W[k*N + n])
__global__ void wtrans(const float* __restrict__ W, unsigned short* __restrict__ Wt,
                       int K, int N) {
  long idx = (long)blockIdx.x * 256 + threadIdx.x;
  if (idx < (long)K * N) {
    long k = idx / N;
    long n = idx - k * N;
    Wt[n * (long)K + k] = f2bf(W[idx]);
  }
}

// ---------------------------------------------------------------------------
// Row LayerNorm -> bf16
__global__ void __launch_bounds__(256)
ln_rows(const float* __restrict__ x, const float* __restrict__ g,
        const float* __restrict__ bb, unsigned short* __restrict__ out, int Dlen) {
  long row = blockIdx.x;
  const float* xr = x + row * (long)Dlen;
  __shared__ float r1[256];
  __shared__ float r2[256];
  int tid = threadIdx.x;
  float s = 0.f, sq = 0.f;
  for (int i = tid; i < Dlen; i += 256) { float v = xr[i]; s += v; sq += v * v; }
  r1[tid] = s; r2[tid] = sq; __syncthreads();
  for (int off = 128; off > 0; off >>= 1) {
    if (tid < off) { r1[tid] += r1[tid + off]; r2[tid] += r2[tid + off]; }
    __syncthreads();
  }
  float mean = r1[0] / Dlen;
  float var  = r2[0] / Dlen - mean * mean;
  float rstd = rsqrtf(var + 1e-5f);
  for (int i = tid; i < Dlen; i += 256) {
    float v = (xr[i] - mean) * rstd * g[i] + bb[i];
    out[row * (long)Dlen + i] = f2bf(v);
  }
}

// ---------------------------------------------------------------------------
// Main tiled WMMA GEMM: C[M x Ntot] = A[M x K](bf16) * Wt[Ntot x K]^T (bf16)
// workgroup = 8 waves, tile 128(M) x 64(N); per-wave 32x32 (2x2 16x16 frags)
// Double-buffered LDS staging via async global->LDS copies (ASYNCcnt).
// MODE: 0=Q(+head softmax)  1=K-motion  2=K-text  3=V-motion  4=V-text  5=Out
template <int MODE>
__global__ void __launch_bounds__(256)
gemm_big(const unsigned short* __restrict__ A, const unsigned short* __restrict__ Wt,
         const float* __restrict__ bias, int M, int K, int Ntot,
         float* __restrict__ fout, unsigned short* __restrict__ bout,
         const float* __restrict__ xres, const int* __restrict__ imask,
         int T, int NT, int Npad) {
  // two 12KB staging buffers (A 128x32 + B 64x32 bf16) + 32KB Cs for MODE 0
  constexpr int SMBYTES = 24576 + ((MODE == 0) ? 32768 : 0);
  __shared__ __attribute__((aligned(16))) unsigned char smem[SMBYTES];
  float* Cs = (float*)(smem + 24576);                  // 128 x 64 f32 (MODE 0)

  const int tid  = threadIdx.x;
  const int lane = tid & 31;
  const int wave = tid >> 5;
  const int wm   = wave >> 1;      // 0..3
  const int wn   = wave & 1;       // 0..1
  const long m0  = (long)blockIdx.y * 128;
  const int  n0  = blockIdx.x * 64;

  v8f acc[2][2];
#pragma unroll
  for (int i = 0; i < 2; ++i)
#pragma unroll
    for (int j = 0; j < 2; ++j)
#pragma unroll
      for (int r = 0; r < 8; ++r) acc[i][j][r] = 0.0f;

  // issue async copies of the (k0) K-slab into buffer p
  auto issue = [&](int k0, int p) {
    unsigned short* Ad = (unsigned short*)(smem + p * 12288);
    unsigned short* Bd = (unsigned short*)(smem + p * 12288 + 8192);
    // A tile (128x32): 512 slots of 8 bf16 (16B)
    for (int s = tid; s < 512; s += 256) {
      int r = s >> 2, ch = s & 3;
      long gm = m0 + r;
      if (gm < M)
        async_b128(Ad + r * 32 + ch * 8, A + gm * (long)K + k0 + ch * 8);
      else
        *(uint4*)(Ad + r * 32 + ch * 8) = make_uint4(0u, 0u, 0u, 0u);
    }
    // B tile (64x32): 256 slots of 16B
    {
      int r = tid >> 2, ch = tid & 3;
      async_b128(Bd + r * 32 + ch * 8, Wt + (long)(n0 + r) * K + k0 + ch * 8);
    }
  };

  issue(0, 0);
  int p = 0;
  for (int k0 = 0; k0 < K; k0 += 32) {
    wait_async0();                 // my copies for buffer p have landed in LDS
    __syncthreads();               // everyone's copies done; other buffer free
    if (k0 + 32 < K) issue(k0 + 32, p ^ 1);   // overlap next slab with compute

    const unsigned short* As = (const unsigned short*)(smem + p * 12288);
    const unsigned short* Bs = (const unsigned short*)(smem + p * 12288 + 8192);
    v16bf a0 = load_a_frag(As, 32, 32 * wm,      0, lane);
    v16bf a1 = load_a_frag(As, 32, 32 * wm + 16, 0, lane);
    v16bf b0 = load_b_frag(Bs, 32, 32 * wn,      0, lane);
    v16bf b1 = load_b_frag(Bs, 32, 32 * wn + 16, 0, lane);
    acc[0][0] = wmma_bf16(a0, b0, acc[0][0]);
    acc[0][1] = wmma_bf16(a0, b1, acc[0][1]);
    acc[1][0] = wmma_bf16(a1, b0, acc[1][0]);
    acc[1][1] = wmma_bf16(a1, b1, acc[1][1]);
    p ^= 1;
  }

  const int c  = lane >> 4;
  const int ln = lane & 15;
#pragma unroll
  for (int mt = 0; mt < 2; ++mt)
#pragma unroll
    for (int nt = 0; nt < 2; ++nt)
#pragma unroll
      for (int r = 0; r < 8; ++r) {
        int row_l = 32 * wm + 16 * mt + 8 * c + r;   // C layout: m = r + 8*(L/16)
        int col_l = 32 * wn + 16 * nt + ln;          //           n = L%16
        long gm = m0 + row_l;
        int  gc = n0 + col_l;
        float val = acc[mt][nt][r] + bias[gc];
        if constexpr (MODE == 0) {
          Cs[row_l * 64 + col_l] = val;
        } else if constexpr (MODE == 1) {            // key-motion -> kT_raw f32
          float sm = (float)imask[gm];
          val += (1.0f - sm) * (-1.0e6f);
          long b = gm / T; long t = gm - b * T;
          int h = gc >> 6, dh = gc & 63;
          fout[(((long)b * 16 + h) * 64 + dh) * (long)Npad + (NT + t)] = val;
        } else if constexpr (MODE == 2) {            // key-text -> kT_raw f32
          if (gm < M) {
            long b = gm / NT; long tt = gm - b * NT;
            float tc = ((imask[b] % 10) > 0) ? 1.0f : 0.0f;
            val += (1.0f - tc) * (-1.0e6f);
            int h = gc >> 6, dh = gc & 63;
            fout[(((long)b * 16 + h) * 64 + dh) * (long)Npad + tt] = val;
          }
        } else if constexpr (MODE == 3) {            // value-motion -> vT bf16
          float sm = (float)imask[gm];
          long b = gm / T; long t = gm - b * T;
          int h = gc >> 6, dh = gc & 63;
          bout[(((long)b * 16 + h) * 64 + dh) * (long)Npad + (NT + t)] = f2bf(val * sm);
        } else if constexpr (MODE == 4) {            // value-text -> vT bf16
          if (gm < M) {
            long b = gm / NT; long tt = gm - b * NT;
            float tc = ((imask[b] % 10) > 0) ? 1.0f : 0.0f;
            int h = gc >> 6, dh = gc & 63;
            bout[(((long)b * 16 + h) * 64 + dh) * (long)Npad + tt] = f2bf(val * tc);
          }
        } else {                                     // MODE 5: out proj + residual
          long idx = gm * (long)Ntot + gc;
          fout[idx] = xres[idx] + val;
        }
      }

  if constexpr (MODE == 0) {                         // head-dim softmax (64 wide)
    __syncthreads();
    if (tid < 128) {
      const float* Crow = Cs + tid * 64;
      float mx = -3.0e38f;
#pragma unroll 8
      for (int j = 0; j < 64; ++j) mx = fmaxf(mx, Crow[j]);
      float s = 0.0f;
#pragma unroll 8
      for (int j = 0; j < 64; ++j) s += __expf(Crow[j] - mx);
      float inv = 1.0f / s;
      unsigned short* orow = bout + (m0 + tid) * (long)Ntot + n0;
#pragma unroll 8
      for (int j = 0; j < 64; ++j) orow[j] = f2bf(__expf(Crow[j] - mx) * inv);
    }
  }
}

// ---------------------------------------------------------------------------
// zero pad columns [Nn, Npad) of vT rows
__global__ void zpad(unsigned short* __restrict__ vT, int Nn, int Npad, long rows) {
  long i = (long)blockIdx.x * 256 + threadIdx.x;
  int pads = Npad - Nn;
  long tot = rows * pads;
  if (i < tot) {
    long r = i / pads;
    int  n = Nn + (int)(i - r * pads);
    vT[r * (long)Npad + n] = 0;
  }
}

// ---------------------------------------------------------------------------
// per-(b,h,dh) softmax over sequence N, write bf16 (pads -> 0)
__global__ void __launch_bounds__(256)
k_softmax(const float* __restrict__ kraw, unsigned short* __restrict__ kout,
          int Nn, int Npad) {
  long row = blockIdx.x;
  const float* xr = kraw + row * (long)Npad;
  unsigned short* orow = kout + row * (long)Npad;
  __shared__ float red[256];
  int tid = threadIdx.x;
  float mx = -3.0e38f;
  for (int i = tid; i < Nn; i += 256) mx = fmaxf(mx, xr[i]);
  red[tid] = mx; __syncthreads();
  for (int off = 128; off > 0; off >>= 1) {
    if (tid < off) red[tid] = fmaxf(red[tid], red[tid + off]);
    __syncthreads();
  }
  mx = red[0]; __syncthreads();
  float s = 0.f;
  for (int i = tid; i < Nn; i += 256) s += __expf(xr[i] - mx);
  red[tid] = s; __syncthreads();
  for (int off = 128; off > 0; off >>= 1) {
    if (tid < off) red[tid] += red[tid + off];
    __syncthreads();
  }
  float inv = 1.0f / red[0];
  for (int i = tid; i < Npad; i += 256)
    orow[i] = (i < Nn) ? f2bf(__expf(xr[i] - mx) * inv) : (unsigned short)0;
}

// ---------------------------------------------------------------------------
// att = kT(64 x Npad) * vT(64 x Npad)^T per (b,h); write transposed attT[l][d]
__global__ void __launch_bounds__(128)
gemm_att(const unsigned short* __restrict__ kT, const unsigned short* __restrict__ vT,
         unsigned short* __restrict__ attT, int Npad) {
  const int bh = blockIdx.x;
  const unsigned short* Am = kT + (long)bh * 64 * Npad;
  const unsigned short* Bt = vT + (long)bh * 64 * Npad;
  const int lane = threadIdx.x & 31;
  const int wave = threadIdx.x >> 5;
  const int wm = wave >> 1, wn = wave & 1;

  v8f acc[2][2];
#pragma unroll
  for (int i = 0; i < 2; ++i)
#pragma unroll
    for (int j = 0; j < 2; ++j)
#pragma unroll
      for (int r = 0; r < 8; ++r) acc[i][j][r] = 0.0f;

  for (int k0 = 0; k0 < Npad; k0 += 32) {
    v16bf a0 = load_a_frag(Am, Npad, 32 * wm,      k0, lane);
    v16bf a1 = load_a_frag(Am, Npad, 32 * wm + 16, k0, lane);
    v16bf b0 = load_b_frag(Bt, Npad, 32 * wn,      k0, lane);
    v16bf b1 = load_b_frag(Bt, Npad, 32 * wn + 16, k0, lane);
    acc[0][0] = wmma_bf16(a0, b0, acc[0][0]);
    acc[0][1] = wmma_bf16(a0, b1, acc[0][1]);
    acc[1][0] = wmma_bf16(a1, b0, acc[1][0]);
    acc[1][1] = wmma_bf16(a1, b1, acc[1][1]);
  }
  const int c = lane >> 4, ln = lane & 15;
#pragma unroll
  for (int mt = 0; mt < 2; ++mt)
#pragma unroll
    for (int nt = 0; nt < 2; ++nt)
#pragma unroll
      for (int r = 0; r < 8; ++r) {
        int m = 32 * wm + 16 * mt + 8 * c + r;     // d
        int n = 32 * wn + 16 * nt + ln;            // l
        attT[(long)bh * 4096 + n * 64 + m] = f2bf(acc[mt][nt][r]);  // [l][d]
      }
}

// ---------------------------------------------------------------------------
// y[t][l] = sum_d q[t][d] * attT[l][d]  per (b,h) ; K = 64
__global__ void __launch_bounds__(256)
gemm_y(const unsigned short* __restrict__ q, const unsigned short* __restrict__ attT,
       float* __restrict__ y, int T) {
  const int bh = blockIdx.x;
  const int b = bh >> 4, h = bh & 15;
  const long m0 = (long)blockIdx.y * 128;
  const unsigned short* Am = q + ((long)b * T) * 1024 + h * 64;  // lda 1024
  const unsigned short* Bt = attT + (long)bh * 4096;             // [l][d], ld 64
  const int lane = threadIdx.x & 31;
  const int wave = threadIdx.x >> 5;
  const int wm = wave >> 1, wn = wave & 1;

  v8f acc[2][2];
#pragma unroll
  for (int i = 0; i < 2; ++i)
#pragma unroll
    for (int j = 0; j < 2; ++j)
#pragma unroll
      for (int r = 0; r < 8; ++r) acc[i][j][r] = 0.0f;

#pragma unroll
  for (int k0 = 0; k0 < 64; k0 += 32) {
    v16bf a0 = load_a_frag(Am, 1024, (int)m0 + 32 * wm,      k0, lane);
    v16bf a1 = load_a_frag(Am, 1024, (int)m0 + 32 * wm + 16, k0, lane);
    v16bf b0 = load_b_frag(Bt, 64, 32 * wn,      k0, lane);
    v16bf b1 = load_b_frag(Bt, 64, 32 * wn + 16, k0, lane);
    acc[0][0] = wmma_bf16(a0, b0, acc[0][0]);
    acc[0][1] = wmma_bf16(a0, b1, acc[0][1]);
    acc[1][0] = wmma_bf16(a1, b0, acc[1][0]);
    acc[1][1] = wmma_bf16(a1, b1, acc[1][1]);
  }
  const int c = lane >> 4, ln = lane & 15;
#pragma unroll
  for (int mt = 0; mt < 2; ++mt)
#pragma unroll
    for (int nt = 0; nt < 2; ++nt)
#pragma unroll
      for (int r = 0; r < 8; ++r) {
        long t = m0 + 32 * wm + 16 * mt + 8 * c + r;
        int  l = 32 * wn + 16 * nt + ln;
        y[((long)b * T + t) * 1024 + h * 64 + l] = acc[mt][nt][r];
      }
}

// ---------------------------------------------------------------------------
__global__ void silu_k(const float* __restrict__ e, float* __restrict__ o, long n) {
  long i = (long)blockIdx.x * 256 + threadIdx.x;
  if (i < n) { float v = e[i]; o[i] = v * (1.0f / (1.0f + __expf(-v))); }
}

// eo[b][j] = sum_i silu(emb)[b][i] * We[i][j] + be[j]
__global__ void __launch_bounds__(256)
emb_gemm(const float* __restrict__ se, const float* __restrict__ We,
         const float* __restrict__ be, float* __restrict__ eo, int TEin, int Nout) {
  long idx = (long)blockIdx.x * 256 + threadIdx.x;
  long b = idx / Nout;
  int  j = (int)(idx - b * Nout);
  const float* sr = se + b * TEin;
  float s = be[j];
  for (int i = 0; i < TEin; ++i) s += sr[i] * We[(long)i * Nout + j];
  eo[idx] = s;
}

// stylization: LN(y)*(1+scale)+shift -> silu -> bf16
__global__ void __launch_bounds__(256)
styl_rows(const float* __restrict__ y, const float* __restrict__ g,
          const float* __restrict__ bb, const float* __restrict__ eo,
          unsigned short* __restrict__ out, int T, int Dlen) {
  long row = blockIdx.x;
  long b = row / T;
  const float* yr = y + row * (long)Dlen;
  const float* sc = eo + b * 2 * (long)Dlen;
  const float* sh = sc + Dlen;
  __shared__ float r1[256];
  __shared__ float r2[256];
  int tid = threadIdx.x;
  float s = 0.f, sq = 0.f;
  for (int i = tid; i < Dlen; i += 256) { float v = yr[i]; s += v; sq += v * v; }
  r1[tid] = s; r2[tid] = sq; __syncthreads();
  for (int off = 128; off > 0; off >>= 1) {
    if (tid < off) { r1[tid] += r1[tid + off]; r2[tid] += r2[tid + off]; }
    __syncthreads();
  }
  float mean = r1[0] / Dlen;
  float var  = r2[0] / Dlen - mean * mean;
  float rstd = rsqrtf(var + 1e-5f);
  for (int i = tid; i < Dlen; i += 256) {
    float v = (yr[i] - mean) * rstd * g[i] + bb[i];
    v = v * (1.0f + sc[i]) + sh[i];
    v = v * (1.0f / (1.0f + __expf(-v)));
    out[row * (long)Dlen + i] = f2bf(v);
  }
}

// ===========================================================================
extern "C" void kernel_launch(void* const* d_in, const int* in_sizes, int n_in,
                              void* d_out, int out_size, void* d_ws, size_t ws_size,
                              hipStream_t stream) {
  (void)in_sizes; (void)n_in; (void)out_size; (void)ws_size;
  const float* x        = (const float*)d_in[0];
  const float* xf       = (const float*)d_in[1];
  const float* emb      = (const float*)d_in[2];
  const int*   src_mask = (const int*)d_in[3];
  const int*   cond     = (const int*)d_in[4];
  const float* norm_g   = (const float*)d_in[5];
  const float* norm_b   = (const float*)d_in[6];
  const float* tnorm_g  = (const float*)d_in[7];
  const float* tnorm_b  = (const float*)d_in[8];
  const float* Wq  = (const float*)d_in[9];   const float* bq  = (const float*)d_in[10];
  const float* Wkt = (const float*)d_in[11];  const float* bkt = (const float*)d_in[12];
  const float* Wvt = (const float*)d_in[13];  const float* bvt = (const float*)d_in[14];
  const float* Wkm = (const float*)d_in[15];  const float* bkm = (const float*)d_in[16];
  const float* Wvm = (const float*)d_in[17];  const float* bvm = (const float*)d_in[18];
  const float* We  = (const float*)d_in[19];  const float* be  = (const float*)d_in[20];
  const float* sb_g = (const float*)d_in[21]; const float* sb_b = (const float*)d_in[22];
  const float* Wo  = (const float*)d_in[23];  const float* bo  = (const float*)d_in[24];
  float* out = (float*)d_out;

  const int B = 32, T = 1024, D = 1024, NT = 77, TL = 768, TE = 2048;
  const int Nn = NT + T;      // 1101
  const int Npad = 1120;      // 35 * 32

  char* wsp = (char*)d_ws;
  size_t off = 0;
  auto carve = [&](size_t bytes) -> void* {
    void* p = wsp + off;
    off = (off + bytes + 255) & ~(size_t)255;
    return p;
  };
  unsigned short* WqT  = (unsigned short*)carve((size_t)D * D * 2);
  unsigned short* WkmT = (unsigned short*)carve((size_t)D * D * 2);
  unsigned short* WvmT = (unsigned short*)carve((size_t)D * D * 2);
  unsigned short* WoT  = (unsigned short*)carve((size_t)D * D * 2);
  unsigned short* WktT = (unsigned short*)carve((size_t)TL * D * 2);
  unsigned short* WvtT = (unsigned short*)carve((size_t)TL * D * 2);
  unsigned short* xn   = (unsigned short*)carve((size_t)B * T * D * 2);
  unsigned short* tn   = (unsigned short*)carve((size_t)B * NT * TL * 2);
  unsigned short* qbf  = (unsigned short*)carve((size_t)B * T * D * 2);
  float*          kraw = (float*)carve((size_t)B * 16 * 64 * Npad * 4);
  unsigned short* kbf  = (unsigned short*)carve((size_t)B * 16 * 64 * Npad * 2);
  unsigned short* vT   = (unsigned short*)carve((size_t)B * 16 * 64 * Npad * 2);
  unsigned short* attT = (unsigned short*)carve((size_t)B * 16 * 64 * 64 * 2);
  float*          semb = (float*)carve((size_t)B * TE * 4);
  float*          eo   = (float*)carve((size_t)B * 2 * D * 4);
  float* y = kraw;               // alias: kraw dead after k_softmax
  unsigned short* hbf = xn;      // alias: xn dead after V/K/Q projections

  // weights -> bf16 transposed
  int nblk = (D * D + 255) / 256;
  wtrans<<<nblk, 256, 0, stream>>>(Wq,  WqT,  D, D);
  wtrans<<<nblk, 256, 0, stream>>>(Wkm, WkmT, D, D);
  wtrans<<<nblk, 256, 0, stream>>>(Wvm, WvmT, D, D);
  wtrans<<<nblk, 256, 0, stream>>>(Wo,  WoT,  D, D);
  nblk = (TL * D + 255) / 256;
  wtrans<<<nblk, 256, 0, stream>>>(Wkt, WktT, TL, D);
  wtrans<<<nblk, 256, 0, stream>>>(Wvt, WvtT, TL, D);

  // layernorms
  ln_rows<<<B * T,  256, 0, stream>>>(x,  norm_g,  norm_b,  xn, D);
  ln_rows<<<B * NT, 256, 0, stream>>>(xf, tnorm_g, tnorm_b, tn, TL);

  dim3 gBig(D / 64, (B * T) / 128);           // (16, 256)
  dim3 gTxt(D / 64, (B * NT + 127) / 128);    // (16, 20)
  gemm_big<0><<<gBig, 256, 0, stream>>>(xn, WqT,  bq,  B * T,  D,  D, nullptr, qbf,
                                        nullptr, nullptr, T, NT, Npad);
  gemm_big<1><<<gBig, 256, 0, stream>>>(xn, WkmT, bkm, B * T,  D,  D, kraw, nullptr,
                                        nullptr, src_mask, T, NT, Npad);
  gemm_big<2><<<gTxt, 256, 0, stream>>>(tn, WktT, bkt, B * NT, TL, D, kraw, nullptr,
                                        nullptr, cond, T, NT, Npad);
  gemm_big<3><<<gBig, 256, 0, stream>>>(xn, WvmT, bvm, B * T,  D,  D, nullptr, vT,
                                        nullptr, src_mask, T, NT, Npad);
  gemm_big<4><<<gTxt, 256, 0, stream>>>(tn, WvtT, bvt, B * NT, TL, D, nullptr, vT,
                                        nullptr, cond, T, NT, Npad);

  {
    long rows = (long)B * 16 * 64;
    long tot = rows * (Npad - Nn);
    zpad<<<(unsigned)((tot + 255) / 256), 256, 0, stream>>>(vT, Nn, Npad, rows);
  }

  k_softmax<<<B * 16 * 64, 256, 0, stream>>>(kraw, kbf, Nn, Npad);

  gemm_att<<<B * 16, 128, 0, stream>>>(kbf, vT, attT, Npad);

  silu_k<<<(B * TE + 255) / 256, 256, 0, stream>>>(emb, semb, (long)B * TE);
  emb_gemm<<<(B * 2 * D) / 256, 256, 0, stream>>>(semb, We, be, eo, TE, 2 * D);

  dim3 gY(B * 16, T / 128);
  gemm_y<<<gY, 256, 0, stream>>>(qbf, attT, y, T);

  styl_rows<<<B * T, 256, 0, stream>>>(y, sb_g, sb_b, eo, hbf, T, D);

  gemm_big<5><<<gBig, 256, 0, stream>>>(hbf, WoT, bo, B * T, D, D, out, nullptr,
                                        x, nullptr, T, NT, Npad);
}
